// GAT_712964571672
// MI455X (gfx1250) — matrix operand order repaired and checked by reading, not compiled, not run on hardware
//
#include <hip/hip_runtime.h>
#include <hip/hip_bf16.h>
#include <math.h>

typedef __attribute__((ext_vector_type(16))) _Float16 v16h;
typedef __attribute__((ext_vector_type(8)))  _Float16 v8h;
typedef __attribute__((ext_vector_type(8)))  float    v8f;

#define Nn   50000
#define Ee   800000
#define Hh   8
#define Cc   16
#define HCc  128
#define EDd  16
#define Gg   1000

// ---------------------------------------------------------------------------
// Weight pre-pack: fp32 [K, Ncols] row-major -> fp16 WMMA B-fragment order.
// B 32x16 fp16 layout (wave32): lane L -> column n = L&15, K-group g = L>>4;
// halves i=0..7 hold K = kc*32 + g*8 + i, i=8..15 hold K = kc*32 + 16 + g*8 + (i-8).
// Out index: ((ct*nkc + kc)*32 + lane)*16 + i  (ct = 16-wide column tile).
// ---------------------------------------------------------------------------
__global__ void pack_b_kernel(const float* __restrict__ W, _Float16* __restrict__ out,
                              int K, int Ncols) {
    int t = blockIdx.x * blockDim.x + threadIdx.x;
    int total = K * Ncols;
    if (t >= total) return;
    int nkc  = K >> 5;
    int i    = t & 15;
    int lane = (t >> 4) & 31;
    int rest = t >> 9;
    int kc   = rest % nkc;
    int ct   = rest / nkc;
    int k = kc * 32 + ((i >> 3) << 4) + ((lane >> 4) << 3) + (i & 7);
    int n = ct * 16 + (lane & 15);
    out[t] = (_Float16)W[(size_t)k * Ncols + n];
}

// ---------------------------------------------------------------------------
// Node linear: xs = xin @ W  ([N,128] x [128,128]).
// One block = 16-row strip; 8 waves, each owning one 16-col tile.
// ---------------------------------------------------------------------------
__global__ void gemm_node_kernel(const float* __restrict__ xin,
                                 const _Float16* __restrict__ Bp,
                                 float* __restrict__ xs) {
    __shared__ __align__(32) _Float16 As[16][128];
    const int lane = threadIdx.x;      // 0..31
    const int wave = threadIdx.y;      // 0..7 -> column tile
    const int row0 = blockIdx.x * 16;
    const int tid  = wave * 32 + lane;

    for (int e = tid; e < 16 * 128; e += 256) {
        int r = e >> 7, c = e & 127;
        As[r][c] = (_Float16)xin[(size_t)(row0 + r) * 128 + c];
    }
    __syncthreads();

    const int g = lane >> 4;           // K half-group
    const int r = lane & 15;           // A row / B column within tile
    v8f acc = {};
    #pragma unroll
    for (int kc = 0; kc < 4; ++kc) {
        const int kb = kc * 32;
        // A 16x32 fp16 layout: lane halves 0..7 -> K=kb+g*8+i, 8..15 -> K=kb+16+g*8+(i-8)
        v8h lo = *(const v8h*)&As[r][kb + g * 8];
        v8h hi = *(const v8h*)&As[r][kb + 16 + g * 8];
        v16h a = __builtin_shufflevector(lo, hi, 0,1,2,3,4,5,6,7,8,9,10,11,12,13,14,15);
        v16h b = *(const v16h*)(Bp + ((size_t)(wave * 4 + kc) * 32 + lane) * 16);
        acc = __builtin_amdgcn_wmma_f32_16x16x32_f16(false, a, false, b,
                                                     (short)0, acc, false, false);
    }
    // C/D: VGPR j -> row j + 8*g, lane -> col
    const int col = wave * 16 + r;
    #pragma unroll
    for (int j = 0; j < 8; ++j)
        xs[(size_t)(row0 + j + 8 * g) * 128 + col] = acc[j];
}

// al_src[n,h] = sum_c xs[n,h,c]*a_s[h,c];  al_dst likewise.
__global__ void node_attn_kernel(const float* __restrict__ xs,
                                 const float* __restrict__ a_s,
                                 const float* __restrict__ a_d,
                                 float* __restrict__ al_src,
                                 float* __restrict__ al_dst) {
    int t = blockIdx.x * blockDim.x + threadIdx.x;
    if (t >= Nn * Hh) return;
    int n = t >> 3, h = t & 7;
    const float* xp = xs + (size_t)n * 128 + h * 16;
    const float* as = a_s + h * 16;
    const float* ad = a_d + h * 16;
    float ss = 0.f, sd = 0.f;
    #pragma unroll
    for (int c = 0; c < 16; ++c) { float v = xp[c]; ss += v * as[c]; sd += v * ad[c]; }
    al_src[t] = ss;
    al_dst[t] = sd;
}

// weff[k,h] = sum_c We[k, h*16+c] * a_e[h,c]   (folds the ea GEMM away)
__global__ void weff_kernel(const float* __restrict__ We,
                            const float* __restrict__ a_e,
                            float* __restrict__ weff) {
    int t = threadIdx.x;
    if (t >= EDd * Hh) return;
    int k = t >> 3, h = t & 7;
    float s = 0.f;
    #pragma unroll
    for (int c = 0; c < 16; ++c) s += We[k * 128 + h * 16 + c] * a_e[h * 16 + c];
    weff[t] = s;
}

// al_e[e,h] = edge_attr[e,:] @ weff[:,h]
__global__ void ale_kernel(const float* __restrict__ ea,
                           const float* __restrict__ weff,
                           float* __restrict__ al_e) {
    int t = blockIdx.x * blockDim.x + threadIdx.x;
    if (t >= Ee * Hh) return;
    int e = t >> 3, h = t & 7;
    const float* ep = ea + (size_t)e * 16;
    float s = 0.f;
    #pragma unroll
    for (int k = 0; k < 16; ++k) s += ep[k] * weff[k * 8 + h];
    al_e[t] = s;
}

__global__ void init_nh_kernel(float* __restrict__ amax, float* __restrict__ denom) {
    int t = blockIdx.x * blockDim.x + threadIdx.x;
    if (t >= Nn * Hh) return;
    amax[t]  = -__builtin_inff();
    denom[t] = 0.f;
}

// Order-preserving float atomic max via int/uint atomics (works for any sign mix).
__device__ __forceinline__ void atomicMaxFloat(float* addr, float val) {
    if (val >= 0.f) atomicMax((int*)addr, __float_as_int(val));
    else            atomicMin((unsigned int*)addr, __float_as_uint(val));
}

// pass 1: alpha = leakyrelu(src+dst+edge, 0.2); segment max over dst
__global__ void edge_alpha_kernel(const int* __restrict__ src, const int* __restrict__ dst,
                                  const float* __restrict__ al_src,
                                  const float* __restrict__ al_dst,
                                  float* __restrict__ alpha,   // in: al_e, out: alpha
                                  float* __restrict__ amax) {
    int t = blockIdx.x * blockDim.x + threadIdx.x;
    if (t >= Ee * Hh) return;
    int e = t >> 3, h = t & 7;
    int s = src[e], d = dst[e];
    float a = al_src[s * 8 + h] + al_dst[d * 8 + h] + alpha[t];
    a = a > 0.f ? a : 0.2f * a;
    alpha[t] = a;
    atomicMaxFloat(&amax[d * 8 + h], a);
}

// pass 2: ex = exp(alpha - amax[dst]); segment sum -> denom
__global__ void edge_exp_kernel(const int* __restrict__ dst,
                                float* __restrict__ alpha,
                                const float* __restrict__ amax,
                                float* __restrict__ denom) {
    int t = blockIdx.x * blockDim.x + threadIdx.x;
    if (t >= Ee * Hh) return;
    int e = t >> 3, h = t & 7;
    int d = dst[e];
    float ex = expf(alpha[t] - amax[d * 8 + h]);
    alpha[t] = ex;
    atomicAdd(&denom[d * 8 + h], ex);
}

__global__ void init_hout_kernel(float* __restrict__ hout, const float* __restrict__ bias) {
    int t = blockIdx.x * blockDim.x + threadIdx.x;
    if (t >= Nn * HCc) return;
    hout[t] = bias[t & 127];
}

// pass 3 (bandwidth-dominant): hout[dst] += xs[src] * alpha_norm ; L2-resident atomics
__global__ void edge_aggr_kernel(const int* __restrict__ src, const int* __restrict__ dst,
                                 const float* __restrict__ ex,
                                 const float* __restrict__ denom,
                                 const float* __restrict__ xs,
                                 float* __restrict__ hout) {
    int t = blockIdx.x * blockDim.x + threadIdx.x;
    if (t >= Ee * Hh) return;
    int e = t >> 3, h = t & 7;
    int s = src[e], d = dst[e];
    float w = ex[t] / (denom[d * 8 + h] + 1e-16f);
    const float4* xp = (const float4*)(xs + (size_t)s * 128 + h * 16);
    float* op = hout + (size_t)d * 128 + h * 16;
    #pragma unroll
    for (int q = 0; q < 4; ++q) {
        float4 v = xp[q];
        atomicAdd(op + 4 * q + 0, v.x * w);
        atomicAdd(op + 4 * q + 1, v.y * w);
        atomicAdd(op + 4 * q + 2, v.z * w);
        atomicAdd(op + 4 * q + 3, v.w * w);
    }
}

__global__ void zero_kernel(float* __restrict__ p, int n) {
    int t = blockIdx.x * blockDim.x + threadIdx.x;
    if (t < n) p[t] = 0.f;
}

__global__ void pool_kernel(const float* __restrict__ h3, const int* __restrict__ batch,
                            float* __restrict__ hpool) {
    int t = blockIdx.x * blockDim.x + threadIdx.x;
    if (t >= Nn * HCc) return;
    int n = t >> 7, j = t & 127;
    atomicAdd(&hpool[(size_t)batch[n] * 128 + j], h3[t]);
}

__global__ void init_out_kernel(float* __restrict__ out, const float* __restrict__ bl2) {
    int t = blockIdx.x * blockDim.x + threadIdx.x;
    if (t < Nn) out[t] = bl2[0];
}

// ---------------------------------------------------------------------------
// Fused MLP head: leakyrelu(cat(h1,h2,h3,pool)@Wl1 + bl1, .01) @ Wl2 + bl2
// WMMA [16,512]x[512,16] per wave; second layer fused in epilogue via
// cross-lane xor-shuffle reduction + one atomicAdd per row per col-tile.
// ---------------------------------------------------------------------------
__global__ void mlp_kernel(const float* __restrict__ h1, const float* __restrict__ h2,
                           const float* __restrict__ h3, const float* __restrict__ hpool,
                           const int* __restrict__ batch,
                           const _Float16* __restrict__ Bp,
                           const float* __restrict__ bl1, const float* __restrict__ Wl2,
                           float* __restrict__ out) {
    __shared__ __align__(32) _Float16 As[16][512];
    const int lane = threadIdx.x;
    const int wave = threadIdx.y;
    const int row0 = blockIdx.x * 16;
    const int tid  = wave * 32 + lane;

    for (int e = tid; e < 16 * 512; e += 256) {
        int r = e >> 9, k = e & 511;
        int row = row0 + r;
        int seg = k >> 7, kk = k & 127;
        float v;
        if      (seg == 0) v = h1[(size_t)row * 128 + kk];
        else if (seg == 1) v = h2[(size_t)row * 128 + kk];
        else if (seg == 2) v = h3[(size_t)row * 128 + kk];
        else               v = hpool[(size_t)batch[row] * 128 + kk];
        As[r][k] = (_Float16)v;
    }
    __syncthreads();

    const int g = lane >> 4;
    const int r = lane & 15;
    const int ct = blockIdx.y * 8 + wave;          // 0..31 column tiles of 512
    v8f acc = {};
    #pragma unroll
    for (int kc = 0; kc < 16; ++kc) {
        const int kb = kc * 32;
        v8h lo = *(const v8h*)&As[r][kb + g * 8];
        v8h hi = *(const v8h*)&As[r][kb + 16 + g * 8];
        v16h a = __builtin_shufflevector(lo, hi, 0,1,2,3,4,5,6,7,8,9,10,11,12,13,14,15);
        v16h b = *(const v16h*)(Bp + ((size_t)(ct * 16 + kc) * 32 + lane) * 16);
        acc = __builtin_amdgcn_wmma_f32_16x16x32_f16(false, a, false, b,
                                                     (short)0, acc, false, false);
    }
    const int col = ct * 16 + r;
    const float bl = bl1[col];
    const float w2 = Wl2[col];
    #pragma unroll
    for (int j = 0; j < 8; ++j) {
        float v = acc[j] + bl;
        v = v > 0.f ? v : 0.01f * v;   // leaky_relu(0.01)
        v *= w2;
        v += __shfl_xor(v, 1, 32);     // reduce over 16 cols (lanes within half-wave)
        v += __shfl_xor(v, 2, 32);
        v += __shfl_xor(v, 4, 32);
        v += __shfl_xor(v, 8, 32);
        if (r == 0) atomicAdd(&out[row0 + j + 8 * g], v);
    }
}

// ---------------------------------------------------------------------------
extern "C" void kernel_launch(void* const* d_in, const int* in_sizes, int n_in,
                              void* d_out, int out_size, void* d_ws, size_t ws_size,
                              hipStream_t stream) {
    (void)in_sizes; (void)n_in; (void)out_size; (void)ws_size;

    const float* x     = (const float*)d_in[0];
    const int*   ei    = (const int*)d_in[1];
    const float* eattr = (const float*)d_in[2];
    const int*   batch = (const int*)d_in[3];
    const int* src = ei;
    const int* dst = ei + Ee;

    const float* W[3], *We[3], *As_[3], *Ad_[3], *Ae_[3], *Bias[3];
    for (int l = 0; l < 3; ++l) {
        int b = 4 + 6 * l;
        W[l]    = (const float*)d_in[b + 0];
        We[l]   = (const float*)d_in[b + 1];
        As_[l]  = (const float*)d_in[b + 2];
        Ad_[l]  = (const float*)d_in[b + 3];
        Ae_[l]  = (const float*)d_in[b + 4];
        Bias[l] = (const float*)d_in[b + 5];
    }
    const float* Wl1 = (const float*)d_in[22];
    const float* bl1 = (const float*)d_in[23];
    const float* Wl2 = (const float*)d_in[24];
    const float* bl2 = (const float*)d_in[25];
    float* out = (float*)d_out;

    // ---- workspace carve-up (256B aligned); total ~134 MB (fits 192MB L2) ----
    char* ws = (char*)d_ws;
    size_t off = 0;
    auto take = [&](size_t bytes) -> char* {
        char* p = ws + off;
        off += (bytes + 255) & ~(size_t)255;
        return p;
    };
    const size_t NB = (size_t)Nn * HCc * sizeof(float);
    _Float16* Wp[3];
    Wp[0] = (_Float16*)take(128 * 128 * sizeof(_Float16));
    Wp[1] = (_Float16*)take(128 * 128 * sizeof(_Float16));
    Wp[2] = (_Float16*)take(128 * 128 * sizeof(_Float16));
    _Float16* Wl1p = (_Float16*)take(512 * 512 * sizeof(_Float16));
    float* weff   = (float*)take(EDd * Hh * sizeof(float));
    float* xs     = (float*)take(NB);
    float* h1     = (float*)take(NB);
    float* h2     = (float*)take(NB);
    float* h3     = (float*)take(NB);
    float* al_src = (float*)take((size_t)Nn * Hh * sizeof(float));
    float* al_dst = (float*)take((size_t)Nn * Hh * sizeof(float));
    float* amax   = (float*)take((size_t)Nn * Hh * sizeof(float));
    float* denom  = (float*)take((size_t)Nn * Hh * sizeof(float));
    float* alphab = (float*)take((size_t)Ee * Hh * sizeof(float));
    float* hpool  = (float*)take((size_t)Gg * HCc * sizeof(float));

    const int TB = 256;
    const int NH  = Nn * Hh;
    const int EH  = Ee * Hh;
    const int NHC = Nn * HCc;

    // weight pre-pack into WMMA fragment order (fp16)
    for (int l = 0; l < 3; ++l)
        pack_b_kernel<<<(128 * 128 + TB - 1) / TB, TB, 0, stream>>>(W[l], Wp[l], 128, 128);
    pack_b_kernel<<<(512 * 512 + TB - 1) / TB, TB, 0, stream>>>(Wl1, Wl1p, 512, 512);

    float* hbuf[3] = { h1, h2, h3 };
    for (int l = 0; l < 3; ++l) {
        const float* xin = (l == 0) ? x : hbuf[l - 1];
        float* hout = hbuf[l];

        gemm_node_kernel<<<dim3(Nn / 16), dim3(32, 8), 0, stream>>>(xin, Wp[l], xs);
        node_attn_kernel<<<(NH + TB - 1) / TB, TB, 0, stream>>>(xs, As_[l], Ad_[l],
                                                                al_src, al_dst);
        weff_kernel<<<1, 128, 0, stream>>>(We[l], Ae_[l], weff);
        ale_kernel<<<(EH + TB - 1) / TB, TB, 0, stream>>>(eattr, weff, alphab);
        init_nh_kernel<<<(NH + TB - 1) / TB, TB, 0, stream>>>(amax, denom);
        edge_alpha_kernel<<<(EH + TB - 1) / TB, TB, 0, stream>>>(src, dst, al_src, al_dst,
                                                                 alphab, amax);
        edge_exp_kernel<<<(EH + TB - 1) / TB, TB, 0, stream>>>(dst, alphab, amax, denom);
        init_hout_kernel<<<(NHC + TB - 1) / TB, TB, 0, stream>>>(hout, Bias[l]);
        edge_aggr_kernel<<<(EH + TB - 1) / TB, TB, 0, stream>>>(src, dst, alphab, denom,
                                                                xs, hout);
    }

    zero_kernel<<<(Gg * HCc + TB - 1) / TB, TB, 0, stream>>>(hpool, Gg * HCc);
    pool_kernel<<<(NHC + TB - 1) / TB, TB, 0, stream>>>(h3, batch, hpool);
    init_out_kernel<<<(Nn + TB - 1) / TB, TB, 0, stream>>>(out, bl2);
    mlp_kernel<<<dim3(Nn / 16, 4), dim3(32, 8), 0, stream>>>(h1, h2, h3, hpool, batch,
                                                             Wl1p, bl1, Wl2, out);
}
